// DeformableHyperCorrelationSCM_36507222016422
// MI455X (gfx1250) — matrix operand order repaired and checked by reading, not compiled, not run on hardware
//
#include <hip/hip_runtime.h>
#include <hip/hip_bf16.h>
#include <cmath>

// ---------------------------------------------------------------------------
// CDNA5 / gfx1250: wave32, WMMA bf16 implicit-GEMM convs.
// A-tiles staged by the Tensor Data Mover (tensor_load_to_lds, TENSORcnt);
// B-tiles staged by async global->LDS copies (ASYNCcnt). Double-buffered LDS.
// ---------------------------------------------------------------------------

typedef __bf16 bf16;
typedef __attribute__((ext_vector_type(16))) __bf16 v16bf;
typedef __attribute__((ext_vector_type(8)))  float  v8f;
typedef __attribute__((ext_vector_type(4)))  unsigned u32x4;
typedef __attribute__((ext_vector_type(8)))  unsigned u32x8;

struct U4 { unsigned x, y, z, w; };
struct F4 { float x, y, z, w; };
union ABu { U4 u[2]; v16bf v; };

constexpr int Bn = 8, Cn = 128, Hn = 128, Wn = 128;
constexpr int HW = Hn * Wn;                 // 16384
constexpr int Mtot = Bn * HW;               // 131072
constexpr int PW = 130;                     // padded width/height (1px halo)
constexpr int PP = PW * PW;                 // padded pixels per image

// ---- CDNA5 async copy primitives ------------------------------------------
__device__ __forceinline__ void async_b128(unsigned lds_off, const bf16* g) {
  asm volatile("global_load_async_to_lds_b128 %0, %1, off"
               :: "v"(lds_off), "v"((unsigned long long)(uintptr_t)g)
               : "memory");
}
template<int N>
__device__ __forceinline__ void wait_asynccnt() {
  asm volatile("s_wait_asynccnt %0" :: "n"(N) : "memory");
}
__device__ __forceinline__ unsigned rfl(unsigned x) {
  return __builtin_amdgcn_readfirstlane(x);
}

// ---------------------------------------------------------------------------
// Implicit-GEMM conv, stride 1.
//  PADIN=true  (KS==3): X is halo-padded NHWC bf16 [B][130][130][ICP]
//  PADIN=false (KS==1): X is dense NHWC bf16 [M][ICP]
//  Wt: [KS*KS][OC][ICP] bf16, Y: dense NHWC fp32 [M][OC].
// WG = 256 thr = 8 waves; tile = 1 image row (128 px) x BN=16*NT cols.
// ---------------------------------------------------------------------------
template<int ICP, int KS, int NT, bool PADIN>
__global__ __launch_bounds__(256) void conv_wmma_kernel(
    const bf16* __restrict__ X, const bf16* __restrict__ Wt,
    float* __restrict__ Y, int OC)
{
  constexpr int BN = NT * 16;
  constexpr int NB = BN / 64;               // async b128s per thread per stage
  constexpr int KITER = ICP / 32;
  constexpr int T = KS * KS * KITER;

  __shared__ bf16 As[2][128 * 32];          // [buf][row][k]   8KB each
  __shared__ bf16 Bs[2][BN * 32];           // [buf][col][k]

  const int tid  = threadIdx.x;
  const int wave = tid >> 5;
  const int lane = tid & 31;
  const int n0   = blockIdx.y * BN;

  const int m_base = blockIdx.x * 128;      // tile == one image row
  const int b = m_base >> 14;
  const int y = (m_base >> 7) & 127;

  const int bn = tid >> 2;                  // B col 0..63
  const int bk = (tid & 3) * 8;             // B k chunk

  const unsigned asBase = (unsigned)(uintptr_t)(&As[0][0]);
  const unsigned bsBase = (unsigned)(uintptr_t)(&Bs[0][0]);

  // ---- TDM D# group 1 for the A tile (constant): 2D tile, 2-byte elements,
  //      dim0 = 32 elems, dim1 = 128 rows, row stride = ICP elems (ISA 8.4) --
  u32x8 g1A;
  g1A[0] = rfl(1u << 16);        // workgroup_mask=0, data_size=1 (2B)
  g1A[1] = rfl(32u << 16);       // tensor_dim0[15:0] = 32
  g1A[2] = rfl(128u << 16);      // tensor_dim0 hi=0 | tensor_dim1[15:0]=128
  g1A[3] = rfl(32u << 16);       // tensor_dim1 hi=0 | tile_dim0=32
  g1A[4] = rfl(128u);            // tile_dim1=128, tile_dim2=0
  g1A[5] = rfl((unsigned)ICP);   // tensor_dim0_stride[31:0] (elements)
  g1A[6] = rfl(0u);              // stride hi | tensor_dim1_stride lo
  g1A[7] = rfl(0u);              // tensor_dim1_stride hi

  v8f acc[NT] = {};

  // wave32 WMMA fragment coords (ISA 7.12.2)
  const int kbaseA = (lane >> 4) * 8;
  const int kbaseB = (lane >> 4) * 8;
  const int arow_off = (wave * 16 + (lane & 15)) * 32;

  auto stage = [&](int s, int t) {
    const int kk  = t / KITER;
    const int ic0 = (t - kk * KITER) * 32;
    // ---- A tile: one TDM op, issued by wave 0 (EXEC ignored by TDM) ----
    if (wave == 0) {
      const bf16* ap;
      if (PADIN) {
        const int kyp = kk / KS, kxp = kk % KS;
        ap = X + ((size_t)((b * PW + y + kyp) * PW) + kxp) * ICP + ic0;
      } else {
        ap = X + (size_t)m_base * ICP + ic0;
      }
      const unsigned long long ga = (unsigned long long)(uintptr_t)ap;
      u32x4 g0;
      g0[0] = rfl(1u);                                   // count=1
      g0[1] = rfl(asBase + (unsigned)s * 8192u);         // lds_addr (bytes)
      g0[2] = rfl((unsigned)ga);                         // global_addr[31:0]
      g0[3] = rfl((unsigned)(ga >> 32) | (2u << 30));    // addr[56:32] | type=2
      asm volatile("tensor_load_to_lds %0, %1" :: "s"(g0), "s"(g1A) : "memory");
    }
    // ---- B tile: async global->LDS b128 copies ----
    const bf16* wp = Wt + ((size_t)kk * OC + n0 + bn) * ICP + ic0 + bk;
    const unsigned boff = bsBase + (unsigned)(s * BN * 32 + bn * 32 + bk) * 2u;
    async_b128(boff, wp);
    if (BN == 128)
      async_b128(boff + 64u * 32u * 2u, wp + (size_t)64 * ICP);
  };

  stage(0, 0);
  int buf = 0;
  for (int t = 0; t < T; ++t) {
    __syncthreads();                        // WAR: all waves done with buf^1
    if (t + 1 < T) {
      stage(buf ^ 1, t + 1);
      __builtin_amdgcn_s_wait_tensorcnt(1); // stage t's TDM done (in-order)
      wait_asynccnt<NB>();                  // stage t's asyncs done (in-order)
    } else {
      __builtin_amdgcn_s_wait_tensorcnt(0);
      wait_asynccnt<0>();
    }
    __syncthreads();                        // stage t visible to all waves

    const bf16* arow = &As[buf][arow_off];
    ABu a;
    a.u[0] = *(const U4*)(arow + kbaseA);
    a.u[1] = *(const U4*)(arow + kbaseA + 16);
    #pragma unroll
    for (int nt = 0; nt < NT; ++nt) {
      const bf16* brow = &Bs[buf][(nt * 16 + (lane & 15)) * 32];
      ABu bb;
      bb.u[0] = *(const U4*)(brow + kbaseB);
      bb.u[1] = *(const U4*)(brow + kbaseB + 16);
      acc[nt] = __builtin_amdgcn_wmma_f32_16x16x32_bf16(
          false, a.v, false, bb.v, (short)0, acc[nt], false, false);
    }
    buf ^= 1;
  }

  // C layout: lane%16 = col; lanes<16 -> M 0-7, lanes>=16 -> M 8-15 (VGPR j)
  const int mg = m_base + wave * 16 + (lane >> 4) * 8;
  const int nc = n0 + (lane & 15);
  #pragma unroll
  for (int nt = 0; nt < NT; ++nt)
    #pragma unroll
    for (int j = 0; j < 8; ++j)
      Y[(size_t)(mg + j) * OC + nc + nt * 16] = acc[nt][j];
}

// ---------------------------------------------------------------------------
// Zero halo border of a padded NHWC bf16 buffer [B][130][130][ICP]
// ---------------------------------------------------------------------------
template<int ICP>
__global__ __launch_bounds__(256) void zero_border_kernel(bf16* Xp)
{
  int i = blockIdx.x * 256 + threadIdx.x;
  if (i >= Bn * PP) return;
  int rr = i % PP;
  int yy = rr / PW, xx = rr - yy * PW;
  if (yy >= 1 && yy <= Hn && xx >= 1 && xx <= Wn) return;
  U4 z{0u, 0u, 0u, 0u};
  U4* p = (U4*)(Xp + (size_t)i * ICP);
  #pragma unroll
  for (int c = 0; c < ICP / 8; ++c) p[c] = z;
}

// ---------------------------------------------------------------------------
// Weight repack: OIHW fp32 -> [KS*KS][OC][ICP] bf16 (zero-padded ic)
// ---------------------------------------------------------------------------
__global__ __launch_bounds__(256) void repack_w_kernel(
    const float* __restrict__ w, bf16* __restrict__ out,
    int OC, int IC, int ICP, int KS)
{
  const int total = KS * KS * OC * ICP;
  for (int i = blockIdx.x * 256 + threadIdx.x; i < total; i += gridDim.x * 256) {
    int kk = i / (OC * ICP);
    int rr = i - kk * (OC * ICP);
    int oc = rr / ICP;
    int ic = rr - oc * ICP;
    float v = 0.f;
    if (ic < IC) v = w[((size_t)(oc * IC + ic) * KS + kk / KS) * KS + (kk % KS)];
    out[i] = (bf16)v;
  }
}

// ---------------------------------------------------------------------------
// Pack inputs: NCHW fp32 -> X0p (concat padded NHWC bf16) + F1/F2 dense NHWC
// ---------------------------------------------------------------------------
__global__ __launch_bounds__(256) void pack_inputs_kernel(
    const float* __restrict__ f1, const float* __restrict__ f2,
    bf16* __restrict__ X0p, float* __restrict__ F1n, float* __restrict__ F2n)
{
  const size_t total = (size_t)Bn * Cn * HW;
  for (size_t i = (size_t)blockIdx.x * 256 + threadIdx.x; i < total;
       i += (size_t)gridDim.x * 256) {
    int p = (int)(i % HW);
    size_t t = i / HW;
    int c  = (int)(t % Cn);
    int bi = (int)(t / Cn);
    int yy = p >> 7, xx = p & 127;
    float a = f1[i], bb = f2[i];
    size_t o = ((size_t)bi * HW + p) * Cn + c;
    F1n[o] = a;
    F2n[o] = bb;
    size_t o2 = ((size_t)((bi * PW + yy + 1) * PW) + xx + 1) * (2 * Cn) + c;
    X0p[o2]      = (bf16)a;
    X0p[o2 + Cn] = (bf16)bb;
  }
}

// ---------------------------------------------------------------------------
// GroupNorm stats: one WG per (b, group). Y dense NHWC fp32 [M, C], 32 groups.
// ---------------------------------------------------------------------------
__global__ __launch_bounds__(256) void gn_stats_kernel(
    const float* __restrict__ Y, float* __restrict__ stats, int C, int CG)
{
  const int bg = blockIdx.x;
  const int bi = bg >> 5;
  const int g  = bg & 31;
  const float* base = Y + (size_t)bi * HW * C + g * CG;
  float s = 0.f, s2 = 0.f;
  for (int p = threadIdx.x; p < HW; p += 256) {
    const float* pp = base + (size_t)p * C;
    for (int c = 0; c < CG; ++c) { float v = pp[c]; s += v; s2 += v * v; }
  }
  __shared__ float r0[256], r1[256];
  r0[threadIdx.x] = s; r1[threadIdx.x] = s2;
  __syncthreads();
  for (int off = 128; off > 0; off >>= 1) {
    if (threadIdx.x < off) {
      r0[threadIdx.x] += r0[threadIdx.x + off];
      r1[threadIdx.x] += r1[threadIdx.x + off];
    }
    __syncthreads();
  }
  if (threadIdx.x == 0) {
    float n = (float)(HW) * (float)CG;
    float mean = r0[0] / n;
    float var  = r1[0] / n - mean * mean;
    stats[bg * 2]     = mean;
    stats[bg * 2 + 1] = rsqrtf(var + 1e-5f);
  }
}

// ---------------------------------------------------------------------------
// GroupNorm apply + act -> bf16. ACT: 1 silu, 2 relu. POUT: padded output.
// ---------------------------------------------------------------------------
template<int ACT, bool POUT>
__global__ __launch_bounds__(256) void gn_apply_kernel(
    const float* __restrict__ Y, bf16* __restrict__ out, int C, int CG, int ostride,
    const float* __restrict__ stats, const float* __restrict__ gamma,
    const float* __restrict__ beta)
{
  const size_t total = (size_t)Mtot * C;
  for (size_t i = (size_t)blockIdx.x * 256 + threadIdx.x; i < total;
       i += (size_t)gridDim.x * 256) {
    int c = (int)(i % C);
    size_t mm = i / C;
    int bi = (int)(mm >> 14);
    int g  = c / CG;
    float mean = stats[(bi * 32 + g) * 2];
    float rstd = stats[(bi * 32 + g) * 2 + 1];
    float v = (Y[i] - mean) * rstd * gamma[c] + beta[c];
    if (ACT == 1) v = v / (1.f + __expf(-v));
    if (ACT == 2) v = fmaxf(v, 0.f);
    size_t o;
    if (POUT) {
      int rem = (int)(mm & 16383);
      int yy = rem >> 7, xx = rem & 127;
      o = ((size_t)((bi * PW + yy + 1) * PW) + xx + 1) * ostride + c;
    } else {
      o = mm * (size_t)ostride + c;
    }
    out[o] = (bf16)v;
  }
}

// ---------------------------------------------------------------------------
// conv3 (64 -> 3, 3x3, pad 1) direct + bias; flow -> d_out, (fx,fy,valid)->ws
// ---------------------------------------------------------------------------
__global__ __launch_bounds__(256) void conv3_pred_kernel(
    const bf16* __restrict__ X2, const float* __restrict__ w,
    const float* __restrict__ bias3, F4* __restrict__ flowv,
    float* __restrict__ out_flow)
{
  __shared__ float wsh[9 * 64 * 3];     // [kidx][c][o]
  for (int i = threadIdx.x; i < 1728; i += 256) {
    int kidx = i / 192; int rr = i - kidx * 192; int c = rr / 3; int o = rr - c * 3;
    wsh[i] = w[((size_t)(o * 64 + c) * 3 + kidx / 3) * 3 + (kidx % 3)];
  }
  __syncthreads();
  const int m = blockIdx.x * 256 + threadIdx.x;
  const int b = m >> 14; const int rem = m & 16383;
  const int y = rem >> 7; const int x = rem & 127;
  float a0 = bias3[0], a1 = bias3[1], a2 = bias3[2];
  for (int ky = -1; ky <= 1; ++ky)
    for (int kx = -1; kx <= 1; ++kx) {
      int yy = y + ky, xx = x + kx;
      if (yy < 0 || yy >= Hn || xx < 0 || xx >= Wn) continue;
      const bf16*  src = X2 + (size_t)(b * HW + yy * Wn + xx) * 64;
      const float* wp  = &wsh[((ky + 1) * 3 + (kx + 1)) * 192];
      #pragma unroll 8
      for (int c = 0; c < 64; ++c) {
        float xv = (float)src[c];
        a0 += xv * wp[c * 3 + 0];
        a1 += xv * wp[c * 3 + 1];
        a2 += xv * wp[c * 3 + 2];
      }
    }
  float validity = 1.f / (1.f + __expf(-a2));
  flowv[m] = F4{a0, a1, validity, 0.f};
  out_flow[(size_t)b * 2 * HW + rem]      = a0;
  out_flow[(size_t)b * 2 * HW + HW + rem] = a1;
}

// ---------------------------------------------------------------------------
// Bilinear warp * validity. One wave per pixel; lane covers 4 channels.
// ---------------------------------------------------------------------------
__global__ __launch_bounds__(256) void warp_kernel(
    const float* __restrict__ F2n, const F4* __restrict__ flowv,
    float* __restrict__ An, float* __restrict__ out_aligned)
{
  const int m = blockIdx.x * 8 + (threadIdx.x >> 5);
  const int lane = threadIdx.x & 31;
  const int b = m >> 14; const int rem = m & 16383;
  const int y = rem >> 7; const int x = rem & 127;
  F4 fv = flowv[m];
  float px = (float)x + fv.x;
  float py = (float)y + fv.y;
  float x0f = floorf(px), y0f = floorf(py);
  float wx = px - x0f, wy = py - y0f;
  int ix = (int)x0f, iy = (int)y0f;
  float cw[4] = {(1.f - wx) * (1.f - wy), wx * (1.f - wy),
                 (1.f - wx) * wy,         wx * wy};
  const int c0 = lane * 4;
  float acc0 = 0.f, acc1 = 0.f, acc2 = 0.f, acc3 = 0.f;
  #pragma unroll
  for (int k = 0; k < 4; ++k) {
    int xi = ix + (k & 1), yi = iy + (k >> 1);
    if (xi < 0 || xi >= Wn || yi < 0 || yi >= Hn) continue;
    F4 vv = *(const F4*)(F2n + (size_t)(b * HW + yi * Wn + xi) * Cn + c0);
    acc0 += cw[k] * vv.x; acc1 += cw[k] * vv.y;
    acc2 += cw[k] * vv.z; acc3 += cw[k] * vv.w;
  }
  F4 ov{acc0 * fv.z, acc1 * fv.z, acc2 * fv.z, acc3 * fv.z};
  *(F4*)(An + (size_t)m * Cn + c0) = ov;
  float* oa = out_aligned + ((size_t)b * Cn + c0) * HW + rem;
  oa[0] = ov.x; oa[HW] = ov.y; oa[2 * HW] = ov.z; oa[3 * HW] = ov.w;
}

// ---------------------------------------------------------------------------
// Cost volume: 81 shifts (ch padded to 96), mean over 128 ch -> bf16
// ---------------------------------------------------------------------------
__global__ __launch_bounds__(256) void cost_volume_kernel(
    const float* __restrict__ F1n, const float* __restrict__ An,
    bf16* __restrict__ CV)
{
  const size_t total = (size_t)Mtot * 96;
  for (size_t i = (size_t)blockIdx.x * 256 + threadIdx.x; i < total;
       i += (size_t)gridDim.x * 256) {
    int k = (int)(i % 96);
    int m = (int)(i / 96);
    if (k >= 81) { CV[i] = (bf16)0.f; continue; }
    int b = m >> 14; int rem = m & 16383;
    int y = rem >> 7; int x = rem & 127;
    int dy = k / 9 - 4, dx = k % 9 - 4;
    int yy = y + dy, xx = x + dx;
    float s = 0.f;
    if (yy >= 0 && yy < Hn && xx >= 0 && xx < Wn) {
      const F4* a  = (const F4*)(F1n + (size_t)m * Cn);
      const F4* bb = (const F4*)(An  + (size_t)(b * HW + yy * Wn + xx) * Cn);
      #pragma unroll 8
      for (int t = 0; t < 32; ++t) {
        F4 av = a[t], bv = bb[t];
        s += av.x * bv.x + av.y * bv.y + av.z * bv.z + av.w * bv.w;
      }
    }
    CV[i] = (bf16)(s * (1.f / 128.f));
  }
}

__global__ __launch_bounds__(256) void abs_diff_kernel(
    const float* __restrict__ F1n, const float* __restrict__ An,
    bf16* __restrict__ D)
{
  const size_t total = (size_t)Mtot * Cn;
  for (size_t i = (size_t)blockIdx.x * 256 + threadIdx.x; i < total;
       i += (size_t)gridDim.x * 256)
    D[i] = (bf16)fabsf(F1n[i] - An[i]);
}

// ---------------------------------------------------------------------------
// lam head: fold lam_w through fus_w2 (linear o linear), then per-pixel dot.
// ---------------------------------------------------------------------------
__global__ void fold_lam_kernel(const float* __restrict__ fus_w2,
                                const float* __restrict__ lam_w,
                                float* __restrict__ vfold)
{
  int t = threadIdx.x;
  float s = 0.f;
  for (int o = 0; o < 128; ++o) s += lam_w[o] * fus_w2[o * 128 + t];
  vfold[t] = s;
}

__global__ __launch_bounds__(256) void lam_kernel(
    const bf16* __restrict__ X4, const float* __restrict__ vfold,
    const float* __restrict__ lam_b, float* __restrict__ out_lam)
{
  const int m = blockIdx.x * 256 + threadIdx.x;
  const bf16* xp = X4 + (size_t)m * 128;
  float s = lam_b[0];
  #pragma unroll 16
  for (int c = 0; c < 128; ++c) s += (float)xp[c] * vfold[c];
  out_lam[m] = 1.f / (1.f + __expf(-s));
}

// ---------------------------------------------------------------------------
// Host orchestration
// ---------------------------------------------------------------------------
extern "C" void kernel_launch(void* const* d_in, const int* in_sizes, int n_in,
                              void* d_out, int out_size, void* d_ws, size_t ws_size,
                              hipStream_t stream) {
  (void)in_sizes; (void)n_in; (void)out_size; (void)ws_size;

  const float* feat1    = (const float*)d_in[0];
  const float* feat2    = (const float*)d_in[1];
  const float* off_w1   = (const float*)d_in[2];
  const float* off_g1   = (const float*)d_in[3];
  const float* off_b1   = (const float*)d_in[4];
  const float* off_w2   = (const float*)d_in[5];
  const float* off_g2   = (const float*)d_in[6];
  const float* off_b2   = (const float*)d_in[7];
  const float* off_w3   = (const float*)d_in[8];
  const float* off_bias3= (const float*)d_in[9];
  const float* corr_w   = (const float*)d_in[10];
  const float* corr_g   = (const float*)d_in[11];
  const float* corr_b   = (const float*)d_in[12];
  const float* diff_w   = (const float*)d_in[13];
  const float* diff_g   = (const float*)d_in[14];
  const float* diff_b   = (const float*)d_in[15];
  const float* fus_w1   = (const float*)d_in[16];
  const float* fus_g1   = (const float*)d_in[17];
  const float* fus_b1   = (const float*)d_in[18];
  const float* fus_w2   = (const float*)d_in[19];
  const float* lam_w    = (const float*)d_in[20];
  const float* lam_b    = (const float*)d_in[21];

  float* out = (float*)d_out;
  float* out_aligned = out;                              // [B,128,H,W]
  float* out_flow    = out + (size_t)Bn * Cn * HW;       // [B,2,H,W]
  float* out_lam     = out_flow + (size_t)Bn * 2 * HW;   // [B,1,H,W]

  // ---- workspace layout (lifetime-safe aliasing) ----
  char* ws = (char*)d_ws;
  constexpr size_t MB = 1ull << 20;
  bf16*  W1    = (bf16*)(ws + 0);
  bf16*  W2    = (bf16*)(ws + 1 * MB);
  bf16*  Wc    = (bf16*)(ws + 2 * MB);
  bf16*  Wd    = (bf16*)(ws + 2 * MB + 256 * 1024);
  bf16*  Wf    = (bf16*)(ws + 3 * MB);
  float* vfold = (float*)(ws + 4 * MB);
  float* stats = (float*)(ws + 4 * MB + 4096);
  F4*    flowv = (F4*)(ws + 5 * MB);                 // 2MB
  bf16*  X0p   = (bf16*)(ws + 8 * MB);               // padded [8][130][130][256] ~69MB
  bf16*  X2    = (bf16*)(ws + 8 * MB);               //   reuse: dense [M,64]
  bf16*  CV    = (bf16*)(ws + 24 * MB);              //   reuse: dense [M,96]
  float* F1n   = (float*)(ws + 80 * MB);             // dense [M,128] f32
  bf16*  X4    = (bf16*)(ws + 80 * MB);              //   reuse after cv/diff
  float* F2n   = (float*)(ws + 144 * MB);            // dense [M,128] f32
  bf16*  DIFF  = (bf16*)(ws + 144 * MB);             //   reuse after warp
  bf16*  X3p   = (bf16*)(ws + 208 * MB);             // padded [8][130][130][128]
  float* Yb    = (float*)(ws + 244 * MB);            // conv out fp32
  bf16*  X1p   = (bf16*)(ws + 308 * MB);             // padded [8][130][130][128]
  float* An    = (float*)(ws + 344 * MB);            // aligned dense NHWC f32

  const dim3 blk(256);
  const int borderGrid = (Bn * PP + 255) / 256;

  // ---- weight repack + lam fold + halo zero (tiny) ----
  repack_w_kernel<<<dim3(1152), blk, 0, stream>>>(off_w1, W1, 128, 256, 256, 3);
  repack_w_kernel<<<dim3(288),  blk, 0, stream>>>(off_w2, W2, 64, 128, 128, 3);
  repack_w_kernel<<<dim3(24),   blk, 0, stream>>>(corr_w, Wc, 64, 81, 96, 1);
  repack_w_kernel<<<dim3(32),   blk, 0, stream>>>(diff_w, Wd, 64, 128, 128, 1);
  repack_w_kernel<<<dim3(576),  blk, 0, stream>>>(fus_w1, Wf, 128, 128, 128, 3);
  fold_lam_kernel<<<dim3(1), dim3(128), 0, stream>>>(fus_w2, lam_w, vfold);
  zero_border_kernel<256><<<dim3(borderGrid), blk, 0, stream>>>(X0p);
  zero_border_kernel<128><<<dim3(borderGrid), blk, 0, stream>>>(X1p);
  zero_border_kernel<128><<<dim3(borderGrid), blk, 0, stream>>>(X3p);

  // ---- pack inputs ----
  pack_inputs_kernel<<<dim3(16384), blk, 0, stream>>>(feat1, feat2, X0p, F1n, F2n);

  // ---- offset branch ----
  conv_wmma_kernel<256, 3, 8, true><<<dim3(1024, 1), blk, 0, stream>>>(X0p, W1, Yb, 128);
  gn_stats_kernel<<<dim3(256), blk, 0, stream>>>(Yb, stats, 128, 4);
  gn_apply_kernel<1, true><<<dim3(16384), blk, 0, stream>>>(Yb, X1p, 128, 4, 128,
                                                            stats, off_g1, off_b1);
  conv_wmma_kernel<128, 3, 4, true><<<dim3(1024, 1), blk, 0, stream>>>(X1p, W2, Yb, 64);
  gn_stats_kernel<<<dim3(256), blk, 0, stream>>>(Yb, stats, 64, 2);
  gn_apply_kernel<1, false><<<dim3(8192), blk, 0, stream>>>(Yb, X2, 64, 2, 64,
                                                            stats, off_g2, off_b2);
  conv3_pred_kernel<<<dim3(512), blk, 0, stream>>>(X2, off_w3, off_bias3,
                                                   flowv, out_flow);

  // ---- warp, cost volume, diff ----
  warp_kernel<<<dim3(16384), blk, 0, stream>>>(F2n, flowv, An, out_aligned);
  cost_volume_kernel<<<dim3(49152), blk, 0, stream>>>(F1n, An, CV);
  abs_diff_kernel<<<dim3(65536), blk, 0, stream>>>(F1n, An, DIFF);

  // ---- corr/diff encoders (1x1) -> ce|de concat (padded X3) ----
  conv_wmma_kernel<96, 1, 4, false><<<dim3(1024, 1), blk, 0, stream>>>(CV, Wc, Yb, 64);
  gn_stats_kernel<<<dim3(256), blk, 0, stream>>>(Yb, stats, 64, 2);
  gn_apply_kernel<2, true><<<dim3(8192), blk, 0, stream>>>(Yb, X3p, 64, 2, 128,
                                                           stats, corr_g, corr_b);
  conv_wmma_kernel<128, 1, 4, false><<<dim3(1024, 1), blk, 0, stream>>>(DIFF, Wd, Yb, 64);
  gn_stats_kernel<<<dim3(256), blk, 0, stream>>>(Yb, stats, 64, 2);
  gn_apply_kernel<2, true><<<dim3(8192), blk, 0, stream>>>(Yb, X3p + 64, 64, 2, 128,
                                                           stats, diff_g, diff_b);

  // ---- fusion conv -> GN+ReLU -> folded lam head ----
  conv_wmma_kernel<128, 3, 8, true><<<dim3(1024, 1), blk, 0, stream>>>(X3p, Wf, Yb, 128);
  gn_stats_kernel<<<dim3(256), blk, 0, stream>>>(Yb, stats, 128, 4);
  gn_apply_kernel<2, false><<<dim3(16384), blk, 0, stream>>>(Yb, X4, 128, 4, 128,
                                                             stats, fus_g1, fus_b1);
  lam_kernel<<<dim3(512), blk, 0, stream>>>(X4, vfold, lam_b, out_lam);
}
